// HAN_VAE_66314295050983
// MI455X (gfx1250) — matrix operand order repaired and checked by reading, not compiled, not run on hardware
//
#include <hip/hip_runtime.h>
#include <hip/hip_bf16.h>

// GatedGraph message passing, MI455X (gfx1250, wave32, bf16 WMMA 16x16x32).
// ws layout (bytes):
//   [0,         7340032)  act  bf16 [14][4096][64]
//   [7340032,   8388608)  h32  f32  [4096][64]
//   [8388608,   8912896)  hbf  bf16 [4096][64]   (row-major h)
//   [8912896,   9281536)  BT   bf16 [3][15][64][64]  (gate, kblock, f, d)
//   [9281536,   9805824)  hT   bf16 [64][4096]   (transposed h, B-frag feed)
// total ~9.8 MB of d_ws required.

#define N_NOTES 4096
#define SIZE    64
#define N_EDGE  14
#define N_ITERS 5

typedef __bf16 bf16x16 __attribute__((ext_vector_type(16)));
typedef float  f32x8   __attribute__((ext_vector_type(8)));
typedef unsigned short u16x8 __attribute__((ext_vector_type(8)));

union Frag16 {
  bf16x16      v;
  u16x8        h[2];
  unsigned int w[8];
};

// round-to-nearest-even f32 -> bf16 (used for h, weights, r*h and act stores)
__device__ __forceinline__ unsigned short f2bf(float f) {
  unsigned u = __float_as_uint(f);
  u += 0x7fffu + ((u >> 16) & 1u);
  return (unsigned short)(u >> 16);
}
__device__ __forceinline__ float sigmoidf_(float x) {
  return 1.0f / (1.0f + __expf(-x));
}

// ---------------------------------------------------------------------------
// k_init: weights -> transposed bf16 stack BT; h32/hbf/hT <- x
// ---------------------------------------------------------------------------
__global__ __launch_bounds__(256) void k_init(
    const float* __restrict__ x,
    const float* __restrict__ wz, const float* __restrict__ wr,
    const float* __restrict__ wh,
    const float* __restrict__ uz, const float* __restrict__ ur,
    const float* __restrict__ uh,
    float* __restrict__ h32, unsigned short* __restrict__ hbf,
    unsigned short* __restrict__ BT, unsigned short* __restrict__ hT) {
  const int HTOT = N_NOTES * SIZE;        // 262144
  const int BTOT = 3 * 15 * 64 * 64;      // 184320
  int idx = blockIdx.x * 256 + threadIdx.x;
  if (idx < HTOT) {
    float v = x[idx];
    unsigned short b = f2bf(v);
    int n = idx >> 6, d = idx & 63;
    h32[idx] = v;
    hbf[idx] = b;
    hT[(size_t)d * N_NOTES + n] = b;
  } else if (idx < HTOT + BTOT) {
    int j   = idx - HTOT;
    int g   = j / 61440;
    int rem = j % 61440;
    int kb  = rem / 4096;
    int r2  = rem % 4096;
    int f   = r2 / 64;
    int d   = r2 % 64;
    const float* W;
    if (g == 0)      W = (kb < 14) ? wz + (size_t)kb * 4096 : uz;
    else if (g == 1) W = (kb < 14) ? wr + (size_t)kb * 4096 : ur;
    else             W = (kb < 14) ? wh + (size_t)kb * 4096 : uh;
    // BT[g][kb][f][d] = W[d][f]   (transposed so B-fragments are contiguous)
    BT[j] = f2bf(W[d * 64 + f]);
  }
}

// ---------------------------------------------------------------------------
// k_act: act[e] = edge[e]^T @ h   (bf16 WMMA, HBM-streaming, LDS-free)
// grid: (32 n-tiles, 14 edge types), 256 threads = 8 independent waves.
// Each wave owns 16 n-columns; A-fragments come straight from global
// (coalesced 64B row segments, bf16 by truncation — exact for {0,1} edges),
// B-fragments come from the hT mirror (two 16B runs per lane, L2-hot).
// ---------------------------------------------------------------------------
__global__ __launch_bounds__(256) void k_act(
    const float* __restrict__ edge,
    const unsigned short* __restrict__ hT,
    unsigned short* __restrict__ act) {
  const int e    = blockIdx.y;
  const int n0   = blockIdx.x * 128;
  const int tid  = threadIdx.x;
  const int lane = tid & 31;
  const int wave = tid >> 5;
  const int lan  = lane & 15;
  const int hsel = lane >> 4;

  // this lane's n column of edge[e] (A matrix row after transpose)
  const float* ap = edge + (size_t)e * N_NOTES * N_NOTES + (n0 + wave * 16 + lan);

  f32x8 acc[4] = {};

#pragma unroll 2
  for (int m0 = 0; m0 < N_NOTES; m0 += 32) {
    // ---- A fragment: K rows {hsel*8..+7, 16+hsel*8..+7} of this column
    unsigned int fa[16];
#pragma unroll
    for (int j = 0; j < 8; ++j)
      fa[j] = __float_as_uint(ap[(size_t)(m0 + hsel * 8 + j) * N_NOTES]);
#pragma unroll
    for (int j = 0; j < 8; ++j)
      fa[8 + j] = __float_as_uint(ap[(size_t)(m0 + 16 + hsel * 8 + j) * N_NOTES]);

    Frag16 A;
#pragma unroll
    for (int i = 0; i < 8; ++i)  // pack two truncated bf16 per VGPR
      A.w[i] = __builtin_amdgcn_perm(fa[2 * i + 1], fa[2 * i], 0x07060302u);

    // ---- B fragments from hT: lane -> (d = dt*16+lan, k = {hsel*8..+7, 16+hsel*8..+7})
    const unsigned short* brow = hT + m0 + hsel * 8;
#pragma unroll
    for (int dt = 0; dt < 4; ++dt) {
      Frag16 B;
      const unsigned short* bp = brow + (size_t)(dt * 16 + lan) * N_NOTES;
      B.h[0] = *(const u16x8*)bp;
      B.h[1] = *(const u16x8*)(bp + 16);
      acc[dt] = __builtin_amdgcn_wmma_f32_16x16x32_bf16(
          false, A.v, false, B.v, (short)0, acc[dt], false, false);
    }
  }

  unsigned short* aBase = act + (size_t)e * N_NOTES * SIZE;
#pragma unroll
  for (int dt = 0; dt < 4; ++dt) {
#pragma unroll
    for (int v = 0; v < 8; ++v) {
      int n = n0 + wave * 16 + v + 8 * hsel;
      int d = dt * 16 + lan;
      aBase[(size_t)n * SIZE + d] = f2bf(acc[dt][v]);
    }
  }
}

// ---------------------------------------------------------------------------
// k_gate: az/ar/ah = [act_0..act_13 | h] @ BT ; GRU gates ; (r*h)@uh ; blend.
// grid: 64 blocks (64 n-rows each) x 128 threads (4 waves, 16 rows/wave)
// ---------------------------------------------------------------------------
__global__ __launch_bounds__(128) void k_gate(
    const unsigned short* __restrict__ act,
    const unsigned short* hbf_in,          // aliases hbf_out
    const float* h32_in,                   // may alias h32_dst
    const unsigned short* __restrict__ BT,
    const float* __restrict__ bz, const float* __restrict__ br,
    const float* __restrict__ bh,
    float* h32_dst, unsigned short* hbf_out, unsigned short* hT_out) {
  const int n0   = blockIdx.x * 64;
  const int tid  = threadIdx.x;
  const int lane = tid & 31;
  const int wave = tid >> 5;
  const int lan  = lane & 15;
  const int hsel = lane >> 4;

  __shared__ float          hs[64 * 68];   // fp32 h tile, pad 64->68
  __shared__ unsigned short rhs[64 * 72];  // bf16 r*h tile, pad 64->72

  {  // stage h tile (32 floats / thread, coalesced)
    int row = tid >> 1;
    int col = (tid & 1) * 32;
    const float4* src = (const float4*)(h32_in + (size_t)(n0 + row) * SIZE + col);
    float4* dst = (float4*)(&hs[row * 68 + col]);
#pragma unroll
    for (int i = 0; i < 8; ++i) dst[i] = src[i];
  }
  __syncthreads();

  f32x8 az[4] = {}, ar[4] = {}, ah[4] = {};
  const int rowg = n0 + wave * 16 + lan;   // global A row for this lane

  for (int kb = 0; kb < 15; ++kb) {
    const unsigned short* Arow =
        (kb < 14) ? act + ((size_t)kb * N_NOTES + rowg) * SIZE
                  : hbf_in + (size_t)rowg * SIZE;
#pragma unroll
    for (int c = 0; c < 2; ++c) {
      Frag16 A;
      const unsigned short* aptr = Arow + c * 32 + hsel * 8;
      A.h[0] = *(const u16x8*)aptr;
      A.h[1] = *(const u16x8*)(aptr + 16);
#pragma unroll
      for (int g = 0; g < 3; ++g) {
        if (g == 2 && kb == 14) continue;  // h-gate uses (r*h)@uh, not h@uh
        const unsigned short* Bbase = BT + (size_t)(g * 15 + kb) * 64 * 64;
        f32x8* acc = (g == 0) ? az : (g == 1) ? ar : ah;
#pragma unroll
        for (int dt = 0; dt < 4; ++dt) {
          Frag16 B;
          const unsigned short* bp =
              Bbase + (dt * 16 + lan) * 64 + c * 32 + hsel * 8;
          B.h[0] = *(const u16x8*)bp;
          B.h[1] = *(const u16x8*)(bp + 16);
          acc[dt] = __builtin_amdgcn_wmma_f32_16x16x32_bf16(
              false, A.v, false, B.v, (short)0, acc[dt], false, false);
        }
      }
    }
  }

  float bzv[4], brv[4], bhv[4];
#pragma unroll
  for (int dt = 0; dt < 4; ++dt) {
    int d = dt * 16 + lan;
    bzv[dt] = bz[d]; brv[dt] = br[d]; bhv[dt] = bh[d];
  }

  // z, r gates; stash r*h in LDS (in A-fragment-friendly row-major layout)
#pragma unroll
  for (int dt = 0; dt < 4; ++dt) {
    int d = dt * 16 + lan;
#pragma unroll
    for (int v = 0; v < 8; ++v) {
      int rl = wave * 16 + v + 8 * hsel;
      float z = sigmoidf_(az[dt][v] + bzv[dt]);
      float r = sigmoidf_(ar[dt][v] + brv[dt]);
      float h = hs[rl * 68 + d];
      rhs[rl * 72 + d] = f2bf(r * h);
      az[dt][v] = z;   // keep gates in the accumulator registers
      ar[dt][v] = r;
    }
  }
  __syncthreads();

  // ah += (r*h) @ uh  (uh stored at BT[2][14])
  {
    const unsigned short* Bbase = BT + (size_t)(2 * 15 + 14) * 64 * 64;
#pragma unroll
    for (int c = 0; c < 2; ++c) {
      Frag16 A;
      const unsigned short* aptr =
          &rhs[(wave * 16 + lan) * 72 + c * 32 + hsel * 8];
      A.h[0] = *(const u16x8*)aptr;
      A.h[1] = *(const u16x8*)(aptr + 16);
#pragma unroll
      for (int dt = 0; dt < 4; ++dt) {
        Frag16 B;
        const unsigned short* bp =
            Bbase + (dt * 16 + lan) * 64 + c * 32 + hsel * 8;
        B.h[0] = *(const u16x8*)bp;
        B.h[1] = *(const u16x8*)(bp + 16);
        ah[dt] = __builtin_amdgcn_wmma_f32_16x16x32_bf16(
            false, A.v, false, B.v, (short)0, ah[dt], false, false);
      }
    }
  }

  // blend: h <- (1-z)*h + r*tanh(ah + bh); store fp32 + bf16 + hT mirrors
#pragma unroll
  for (int dt = 0; dt < 4; ++dt) {
    int d = dt * 16 + lan;
#pragma unroll
    for (int v = 0; v < 8; ++v) {
      int rl = wave * 16 + v + 8 * hsel;
      float z  = az[dt][v];
      float r  = ar[dt][v];
      float h  = hs[rl * 68 + d];
      float ht = tanhf(ah[dt][v] + bhv[dt]);
      float hn = (1.0f - z) * h + r * ht;
      unsigned short hb = f2bf(hn);
      size_t gi = (size_t)(n0 + rl) * SIZE + d;
      h32_dst[gi] = hn;
      hbf_out[gi] = hb;
      hT_out[(size_t)d * N_NOTES + (n0 + rl)] = hb;
    }
  }
}

// ---------------------------------------------------------------------------
extern "C" void kernel_launch(void* const* d_in, const int* in_sizes, int n_in,
                              void* d_out, int out_size, void* d_ws,
                              size_t ws_size, hipStream_t stream) {
  const float* x    = (const float*)d_in[0];
  const float* edge = (const float*)d_in[1];
  const float* wz   = (const float*)d_in[2];
  const float* wr   = (const float*)d_in[3];
  const float* wh   = (const float*)d_in[4];
  const float* uz   = (const float*)d_in[5];
  const float* ur   = (const float*)d_in[6];
  const float* uh   = (const float*)d_in[7];
  const float* bz   = (const float*)d_in[8];
  const float* br   = (const float*)d_in[9];
  const float* bh   = (const float*)d_in[10];
  float* out = (float*)d_out;

  char* ws = (char*)d_ws;
  unsigned short* act = (unsigned short*)(ws);
  float*          h32 = (float*)(ws + 7340032);
  unsigned short* hbf = (unsigned short*)(ws + 8388608);
  unsigned short* BT  = (unsigned short*)(ws + 8912896);
  unsigned short* hT  = (unsigned short*)(ws + 9281536);

  const int initTot = N_NOTES * SIZE + 3 * 15 * 64 * 64;
  k_init<<<(initTot + 255) / 256, 256, 0, stream>>>(x, wz, wr, wh, uz, ur, uh,
                                                    h32, hbf, BT, hT);
  for (int it = 0; it < N_ITERS; ++it) {
    k_act<<<dim3(32, 14), 256, 0, stream>>>(edge, hT, act);
    float* dst32 = (it == N_ITERS - 1) ? out : h32;
    k_gate<<<64, 128, 0, stream>>>(act, hbf, h32, BT, bz, br, bh, dst32, hbf,
                                   hT);
  }
  (void)in_sizes; (void)n_in; (void)out_size; (void)ws_size;
}
